// QuantMatMul_57114475102679
// MI455X (gfx1250) — 8-bit matrix operand order repaired and checked by reading, compiled, not run on hardware
//
#include <hip/hip_runtime.h>
#include <hip/hip_bf16.h>

typedef int   v8i_t __attribute__((ext_vector_type(8)));
typedef float v8f_t __attribute__((ext_vector_type(8)));

#define ASTR 80
#define BSTR 80

__device__ __forceinline__ float qlevel(float x, float lo, float up, float s, float off) {
    return rintf(__builtin_amdgcn_fmed3f(x, lo, up) * s + off);
}

__device__ __forceinline__ unsigned fp8_byte(float f) {
    int c = (__float_as_int(f) - 0x3C000000) >> 20;
    return (unsigned)(c > 0 ? c : 0);
}

__global__ __launch_bounds__(256)
void QuantMatMul_57114475102679_kernel(const float* __restrict__ A,
                                       const float* __restrict__ B,
                                       const float* __restrict__ aL,
                                       const float* __restrict__ aU,
                                       const float* __restrict__ bL,
                                       const float* __restrict__ bU,
                                       float* __restrict__ O) {
    __shared__ unsigned char Ash[256 * ASTR];
    __shared__ unsigned char Bsh[256 * BSTR];
    __shared__ float rowsumA[256];
    __shared__ float colsumB[256];

    const int tid   = threadIdx.x;
    const int batch = blockIdx.x;

    const float alo = aL[0], aup = aU[0];
    const float blo = bL[0], bup = bU[0];
    const float sA   = (aup - alo) * (1.0f / 15.0f);
    const float sB   = (bup - blo) * (1.0f / 15.0f);
    const float ainv = 15.0f / (aup - alo);
    const float binv = 15.0f / (bup - blo);
    const float aoff = -alo * ainv;
    const float boff = -blo * binv;

    {
        const float4* ap = (const float4*)(A + (size_t)batch * 16384 + (size_t)tid * 64);
        float rsum = 0.0f;
        #pragma unroll
        for (int g = 0; g < 4; ++g) {
            unsigned w[4];
            #pragma unroll
            for (int j = 0; j < 4; ++j) {
                float4 x = ap[g * 4 + j];
                float f0 = qlevel(x.x, alo, aup, ainv, aoff);
                float f1 = qlevel(x.y, alo, aup, ainv, aoff);
                float f2 = qlevel(x.z, alo, aup, ainv, aoff);
                float f3 = qlevel(x.w, alo, aup, ainv, aoff);
                rsum += (f0 + f1) + (f2 + f3);
                w[j] = fp8_byte(f0) | (fp8_byte(f1) << 8) |
                       (fp8_byte(f2) << 16) | (fp8_byte(f3) << 24);
            }
            *(uint4*)(Ash + tid * ASTR + g * 16) = make_uint4(w[0], w[1], w[2], w[3]);
        }
        rowsumA[tid] = rsum;
    }

    {
        const float* bp = B + (size_t)batch * 16384 + tid;
        float csum = 0.0f;
        #pragma unroll 4
        for (int g = 0; g < 16; ++g) {
            unsigned w = 0;
            #pragma unroll
            for (int j = 0; j < 4; ++j) {
                float f = qlevel(bp[(size_t)(g * 4 + j) * 256], blo, bup, binv, boff);
                csum += f;
                w |= fp8_byte(f) << (j * 8);
            }
            *(unsigned*)(Bsh + tid * BSTR + g * 4) = w;
        }
        colsumB[tid] = csum;
    }

    __syncthreads();

    const int wave = tid >> 5;
    const int lane = tid & 31;
    const int hi = lane >> 4;
    const int ln = lane & 15;
    const int rb = wave * 32;

    v8i_t af[2];
    #pragma unroll
    for (int mb = 0; mb < 2; ++mb) {
        const unsigned char* ap = Ash + (rb + mb * 16 + ln) * ASTR + hi * 16;
        af[mb][0] = *(const int*)(ap + 0);
        af[mb][1] = *(const int*)(ap + 4);
        af[mb][2] = *(const int*)(ap + 8);
        af[mb][3] = *(const int*)(ap + 12);
        af[mb][4] = *(const int*)(ap + 32);
        af[mb][5] = *(const int*)(ap + 36);
        af[mb][6] = *(const int*)(ap + 40);
        af[mb][7] = *(const int*)(ap + 44);
    }

    const float sAB = sA * sB;
    const float cAr = sA * blo;
    const float cBc = sB * alo;
    const float kb  = 64.0f * alo * blo;
    float rterm[2][8];
    #pragma unroll
    for (int mb = 0; mb < 2; ++mb) {
        float4 r0 = *(const float4*)(&rowsumA[rb + mb * 16 + hi * 8]);
        float4 r1 = *(const float4*)(&rowsumA[rb + mb * 16 + hi * 8 + 4]);
        rterm[mb][0] = cAr * r0.x + kb;  rterm[mb][1] = cAr * r0.y + kb;
        rterm[mb][2] = cAr * r0.z + kb;  rterm[mb][3] = cAr * r0.w + kb;
        rterm[mb][4] = cAr * r1.x + kb;  rterm[mb][5] = cAr * r1.y + kb;
        rterm[mb][6] = cAr * r1.z + kb;  rterm[mb][7] = cAr * r1.w + kb;
    }

    #pragma unroll 2
    for (int s = 0; s < 16; ++s) {
        const int cc = s * 16 + ln;
        const unsigned char* bp = Bsh + cc * BSTR + hi * 16;
        int4 b0 = *(const int4*)(bp);
        int4 b1 = *(const int4*)(bp + 32);
        v8i_t bf = {b0.x, b0.y, b0.z, b0.w, b1.x, b1.y, b1.z, b1.w};

        const float cterm = cBc * colsumB[cc];

        #pragma unroll
        for (int mb = 0; mb < 2; ++mb) {
            v8f_t acc = {};
            acc = __builtin_amdgcn_wmma_f32_16x16x64_fp8_fp8(af[mb], bf,
                                                             (short)0, acc, false, false);
            float* op = O + (size_t)batch * 65536
                          + (size_t)(rb + mb * 16 + hi * 8) * 256 + cc;
            #pragma unroll
            for (int v = 0; v < 8; ++v)
                op[(size_t)v * 256] = sAB * acc[v] + rterm[mb][v] + cterm;
        }
    }
}

extern "C" void kernel_launch(void* const* d_in, const int* in_sizes, int n_in,
                              void* d_out, int out_size, void* d_ws, size_t ws_size,
                              hipStream_t stream) {
    const float* A  = (const float*)d_in[0];
    const float* B  = (const float*)d_in[1];
    const float* aL = (const float*)d_in[2];
    const float* aU = (const float*)d_in[3];
    const float* bL = (const float*)d_in[4];
    const float* bU = (const float*)d_in[5];
    float* O = (float*)d_out;

    dim3 grid(1024);
    dim3 block(256);
    QuantMatMul_57114475102679_kernel<<<grid, block, 0, stream>>>(A, B, aL, aU, bL, bU, O);
}
